// Quamba2ChunkScan_40243843564248
// MI455X (gfx1250) — 16-bit and 8-bit matrix operand order repaired and checked by reading, compiled, not run on hardware
//
#include <hip/hip_runtime.h>
#include <stdint.h>

#define B_SZ 2
#define LSEQ 2048
#define NH   32
#define PD   64
#define NS   128
#define CSZ  256
#define NCH  (LSEQ / CSZ)

typedef __attribute__((ext_vector_type(16))) _Float16 v16h;
typedef __attribute__((ext_vector_type(8)))  _Float16 v8h;
typedef __attribute__((ext_vector_type(8)))  float    v8f;
typedef __attribute__((ext_vector_type(8)))  int      v8i;
typedef __attribute__((ext_vector_type(4)))  int      v4i;

__device__ __forceinline__ v16h frag_a_f16(const _Float16* tile, int lane, int ldk) {
  int m = lane & 15, half = lane >> 4;
  const _Float16* p = tile + m * ldk + half * 16;
  v8h lo = *(const v8h*)(p);
  v8h hi = *(const v8h*)(p + 8);
  v16h r;
#pragma unroll
  for (int i = 0; i < 8; ++i) { r[i] = lo[i]; r[i + 8] = hi[i]; }
  return r;
}
__device__ __forceinline__ v16h frag_b_f16(const _Float16* tile, int lane, int ldk) {
  int n = lane & 15, half = lane >> 4;
  return *(const v16h*)(tile + n * ldk + half * 16);
}
__device__ __forceinline__ v8i frag_a_i8(const int8_t* tile, int lane, int ldk) {
  int m = lane & 15, half = lane >> 4;
  const int8_t* p = tile + m * ldk + half * 16;
  union { v8i v; v4i q[2]; } u;
  u.q[0] = *(const v4i*)(p);
  u.q[1] = *(const v4i*)(p + 32);
  return u.v;
}
__device__ __forceinline__ v8i frag_b_i8(const int8_t* tile, int lane, int ldk) {
  int n = lane & 15, half = lane >> 4;
  const int8_t* p = tile + n * ldk + half * 16;
  union { v8i v; v4i q[2]; } u;
  u.q[0] = *(const v4i*)(p);
  u.q[1] = *(const v4i*)(p + 32);
  return u.v;
}

__device__ __forceinline__ float softplusf(float t) {
  return (t > 20.f) ? t : log1pf(expf(t));
}

__global__ void quamba_chunk_states(
    const int8_t* __restrict__ x, const int8_t* __restrict__ dt,
    const int8_t* __restrict__ Bq, const int8_t* __restrict__ qA,
    const float* __restrict__ dt_bias,
    const float* __restrict__ pAsc, const float* __restrict__ pDTsc,
    const float* __restrict__ pXsc, const float* __restrict__ pBsc,
    float* __restrict__ ws_states, float* __restrict__ ws_cum,
    float* __restrict__ ws_dtf) {
  extern __shared__ char smem[];
  _Float16* sBT  = (_Float16*)(smem);
  _Float16* sXwT = (_Float16*)(smem + 65536);
  float*    sCum = (float*)(smem + 98304);
  float*    sDtw = sCum + CSZ;

  const int tid = threadIdx.x;
  const int idx = blockIdx.x;
  const int h = idx & (NH - 1);
  const int z = (idx >> 5) & (NCH - 1);
  const int bb = idx >> 8;
  const int bzh = idx;

  const float A    = -expf((float)qA[h] * (*pAsc));
  const float bias = dt_bias[h];
  const float dtsc = *pDTsc;

  {
    int8_t dq = dt[(((size_t)bb * LSEQ) + (size_t)z * CSZ + tid) * NH + h];
    float sp = softplusf((float)dq * dtsc + bias);
    sCum[tid] = sp * A;
    sDtw[tid] = sp;
    __syncthreads();
#pragma unroll
    for (int off = 1; off < CSZ; off <<= 1) {
      float v = (tid >= off) ? sCum[tid - off] : 0.f;
      __syncthreads();
      sCum[tid] += v;
      __syncthreads();
    }
    float cum = sCum[tid];
    float cumlast = sCum[CSZ - 1];
    float dtf = sDtw[tid];
    ws_cum[(size_t)bzh * CSZ + tid] = cum;
    ws_dtf[(size_t)bzh * CSZ + tid] = dtf;
    sDtw[tid] = dtf * expf(cumlast - cum);
    __syncthreads();
  }

  const int8_t* xc = x + ((((size_t)bb * LSEQ) + (size_t)z * CSZ) * NH + h) * PD;
#pragma unroll
  for (int it = 0; it < 16; ++it) {
    int i = it * 256 + tid;
    int s = i >> 4, p4 = (i & 15) * 4;
    int32_t w4 = *(const int32_t*)(xc + (size_t)s * NH * PD + p4);
    float wgt = sDtw[s];
#pragma unroll
    for (int j = 0; j < 4; ++j)
      sXwT[(p4 + j) * CSZ + s] = (_Float16)((float)(int8_t)(w4 >> (8 * j)) * wgt);
  }
  const int8_t* bc = Bq + (((size_t)bb * LSEQ) + (size_t)z * CSZ) * NS;
#pragma unroll
  for (int it = 0; it < 32; ++it) {
    int i = it * 256 + tid;
    int s = i >> 5, n4 = (i & 31) * 4;
    int32_t w4 = *(const int32_t*)(bc + (size_t)s * NS + n4);
#pragma unroll
    for (int j = 0; j < 4; ++j)
      sBT[(n4 + j) * CSZ + s] = (_Float16)(float)(int8_t)(w4 >> (8 * j));
  }
  __syncthreads();

  const int lane = tid & 31, wave = tid >> 5;
  const int mt = wave >> 1, nt0 = (wave & 1) * 4;
  const float outscale = (*pXsc) * (*pBsc);
  float* outp = ws_states + (size_t)bzh * PD * NS;
  for (int nt = nt0; nt < nt0 + 4; ++nt) {
    v8f acc = {};
#pragma unroll
    for (int k0 = 0; k0 < CSZ; k0 += 32) {
      v16h a = frag_a_f16(sXwT + mt * 16 * CSZ + k0, lane, CSZ);
      v16h b = frag_b_f16(sBT + nt * 16 * CSZ + k0, lane, CSZ);
      acc = __builtin_amdgcn_wmma_f32_16x16x32_f16(false, a, false, b,
                                                   (short)0, acc, false, false);
    }
    int half = lane >> 4, col = lane & 15;
#pragma unroll
    for (int r = 0; r < 8; ++r) {
      int m = mt * 16 + r + 8 * half;
      int n = nt * 16 + col;
      outp[m * NS + n] = acc[r] * outscale;
    }
  }
}

__global__ void quamba_chunk_scan_states(float* __restrict__ ws_states,
                                         const float* __restrict__ ws_cum) {
  const int tid = threadIdx.x;
  const int h = blockIdx.x & (NH - 1);
  const int bb = blockIdx.x >> 5;
  float carry[32];
#pragma unroll
  for (int j = 0; j < 32; ++j) carry[j] = 0.f;
  for (int z = 0; z < NCH; ++z) {
    int bzh = (bb * NCH + z) * NH + h;
    float d = expf(ws_cum[(size_t)bzh * CSZ + CSZ - 1]);
    float* st = ws_states + (size_t)bzh * PD * NS;
#pragma unroll
    for (int j = 0; j < 32; ++j) {
      int i = j * 256 + tid;
      float s = st[i];
      st[i] = carry[j];
      carry[j] = d * carry[j] + s;
    }
  }
}

__global__ void quamba_chunk_out(
    const int8_t* __restrict__ x, const int8_t* __restrict__ Bq,
    const int8_t* __restrict__ Cq, const int8_t* __restrict__ qD,
    const float* __restrict__ pDsc, const float* __restrict__ pXsc,
    const float* __restrict__ pBsc, const float* __restrict__ pCsc,
    const float* __restrict__ ws_states, const float* __restrict__ ws_cum,
    const float* __restrict__ ws_dtf, float* __restrict__ out) {
  extern __shared__ char smem[];
  int8_t*   sB8  = (int8_t*)(smem);
  int8_t*   sC8  = (int8_t*)(smem + 32768);
  _Float16* sCf  = (_Float16*)(smem + 40960);
  _Float16* sXT  = (_Float16*)(smem + 57344);
  _Float16* sPT  = (_Float16*)(smem + 90112);
  _Float16* sSc  = (_Float16*)(smem + 106496);
  float*    sCum = (float*)(smem + 139264);
  float*    sDtf = sCum + CSZ;

  const int tid = threadIdx.x;
  const int idx = blockIdx.x;
  const int lb = idx & 3;
  const int h  = (idx >> 2) & (NH - 1);
  const int z  = (idx >> 7) & (NCH - 1);
  const int bb = idx >> 10;
  const int bzh = (bb * NCH + z) * NH + h;

  sCum[tid] = ws_cum[(size_t)bzh * CSZ + tid];
  sDtf[tid] = ws_dtf[(size_t)bzh * CSZ + tid];

  {
    const int* src = (const int*)(Bq + (((size_t)bb * LSEQ) + (size_t)z * CSZ) * NS);
    int* dst = (int*)sB8;
#pragma unroll
    for (int it = 0; it < 32; ++it) dst[it * 256 + tid] = src[it * 256 + tid];
  }
  {
    const int* src = (const int*)(Cq + (((size_t)bb * LSEQ) + (size_t)z * CSZ +
                                        (size_t)lb * 64) * NS);
    int* dst = (int*)sC8;
#pragma unroll
    for (int it = 0; it < 8; ++it) {
      int i = it * 256 + tid;
      int w4 = src[i];
      dst[i] = w4;
#pragma unroll
      for (int j = 0; j < 4; ++j)
        sCf[i * 4 + j] = (_Float16)(float)(int8_t)(w4 >> (8 * j));
    }
  }
  {
    const int8_t* xc = x + ((((size_t)bb * LSEQ) + (size_t)z * CSZ) * NH + h) * PD;
#pragma unroll
    for (int it = 0; it < 16; ++it) {
      int i = it * 256 + tid;
      int s = i >> 4, p4 = (i & 15) * 4;
      int32_t w4 = *(const int32_t*)(xc + (size_t)s * NH * PD + p4);
#pragma unroll
      for (int j = 0; j < 4; ++j)
        sXT[(p4 + j) * CSZ + s] = (_Float16)(float)(int8_t)(w4 >> (8 * j));
    }
  }
  {
    const float* prev = ws_states + (size_t)bzh * PD * NS;
#pragma unroll
    for (int it = 0; it < 32; ++it) {
      int i = it * 256 + tid;
      sPT[i] = (_Float16)prev[i];
    }
  }
  __syncthreads();

  const int lane = tid & 31, wave = tid >> 5;
  const float sIBX = (*pCsc) * (*pBsc) * (*pXsc);

  for (int t = wave * 8; t < wave * 8 + 8; ++t) {
    int mt = t >> 4, st = t & 15;
    v8i acc = {};
#pragma unroll
    for (int k0 = 0; k0 < NS; k0 += 64) {
      v8i a = frag_a_i8(sC8 + mt * 16 * NS + k0, lane, NS);
      v8i b = frag_b_i8(sB8 + st * 16 * NS + k0, lane, NS);
      acc = __builtin_amdgcn_wmma_i32_16x16x64_iu8(true, a, true, b,
                                                   acc, false, false);
    }
    int half = lane >> 4, col = lane & 15;
#pragma unroll
    for (int r = 0; r < 8; ++r) {
      int m = mt * 16 + r + 8 * half;
      int l = lb * 64 + m;
      int s = st * 16 + col;
      float wgt = (s <= l) ? expf(sCum[l] - sCum[s]) * sDtf[s] * sIBX : 0.f;
      sSc[m * CSZ + s] = (_Float16)((float)acc[r] * wgt);
    }
  }
  __syncthreads();

  const float Df  = (float)qD[h] * (*pDsc);
  const float xsc = *pXsc;
  const float csc = *pCsc;
  for (int t = wave; t < 16; t += 8) {
    int mt = t >> 2, pt = t & 3;
    v8f ai = {};
#pragma unroll
    for (int k0 = 0; k0 < CSZ; k0 += 32) {
      v16h a = frag_a_f16(sSc + mt * 16 * CSZ + k0, lane, CSZ);
      v16h b = frag_b_f16(sXT + pt * 16 * CSZ + k0, lane, CSZ);
      ai = __builtin_amdgcn_wmma_f32_16x16x32_f16(false, a, false, b,
                                                  (short)0, ai, false, false);
    }
    v8f ae = {};
#pragma unroll
    for (int k0 = 0; k0 < NS; k0 += 32) {
      v16h a = frag_a_f16(sCf + mt * 16 * NS + k0, lane, NS);
      v16h b = frag_b_f16(sPT + pt * 16 * NS + k0, lane, NS);
      ae = __builtin_amdgcn_wmma_f32_16x16x32_f16(false, a, false, b,
                                                  (short)0, ae, false, false);
    }
    int half = lane >> 4, col = lane & 15;
#pragma unroll
    for (int r = 0; r < 8; ++r) {
      int m = mt * 16 + r + 8 * half;
      int l = lb * 64 + m;
      int p = pt * 16 + col;
      float xraw = (float)sXT[p * CSZ + l];
      float y = ai[r] + ae[r] * csc * expf(sCum[l]) + xraw * xsc * Df;
      out[((((size_t)bb * LSEQ) + (size_t)z * CSZ + l) * NH + h) * PD + p] = y;
    }
  }
}

extern "C" void kernel_launch(void* const* d_in, const int* in_sizes, int n_in,
                              void* d_out, int out_size, void* d_ws, size_t ws_size,
                              hipStream_t stream) {
  (void)in_sizes; (void)n_in; (void)out_size; (void)ws_size;
  const int8_t* x   = (const int8_t*)d_in[0];
  const int8_t* dt  = (const int8_t*)d_in[1];
  const int8_t* Bq  = (const int8_t*)d_in[2];
  const int8_t* Cq  = (const int8_t*)d_in[3];
  const int8_t* qA  = (const int8_t*)d_in[4];
  const int8_t* qD  = (const int8_t*)d_in[5];
  const float* dt_bias = (const float*)d_in[6];
  const float* pAsc  = (const float*)d_in[7];
  const float* pDsc  = (const float*)d_in[8];
  const float* pDTsc = (const float*)d_in[9];
  const float* pXsc  = (const float*)d_in[10];
  const float* pBsc  = (const float*)d_in[11];
  const float* pCsc  = (const float*)d_in[12];
  float* out = (float*)d_out;

  float* ws_states = (float*)d_ws;
  float* ws_cum = ws_states + (size_t)B_SZ * NCH * NH * PD * NS;
  float* ws_dtf = ws_cum + (size_t)B_SZ * NCH * NH * CSZ;

  const size_t lds1 = 65536 + 32768 + 2 * CSZ * sizeof(float);
  const size_t lds3 = 139264 + 2 * CSZ * sizeof(float);

  quamba_chunk_states<<<B_SZ * NCH * NH, 256, lds1, stream>>>(
      x, dt, Bq, qA, dt_bias, pAsc, pDTsc, pXsc, pBsc,
      ws_states, ws_cum, ws_dtf);

  quamba_chunk_scan_states<<<B_SZ * NH, 256, 0, stream>>>(ws_states, ws_cum);

  quamba_chunk_out<<<B_SZ * NCH * NH * 4, 256, lds3, stream>>>(
      x, Bq, Cq, qD, pDsc, pXsc, pBsc, pCsc,
      ws_states, ws_cum, ws_dtf, out);
}
